// net_81930796138744
// MI455X (gfx1250) — matrix operand order repaired and checked by reading, not compiled, not run on hardware
//
#include <hip/hip_runtime.h>

// ---------------- problem geometry ----------------
#define K_D   64
#define HID   100
#define OUTN  384

typedef __attribute__((ext_vector_type(16))) _Float16 v16h;
typedef __attribute__((ext_vector_type(8)))  _Float16 v8h;
typedef __attribute__((ext_vector_type(4)))  _Float16 v4h;
typedef __attribute__((ext_vector_type(8)))  float    v8f;
typedef __attribute__((ext_vector_type(4)))  float    v4f;

// ---------------- packed-weight geometry (f16 WMMA B-fragments in d_ws) ----
#define KS1 2              // GEMM1: K=64  -> 2 ksteps of 32
#define NT1 7              // GEMM1: N=100 -> 7 tiles of 16 (zero padded)
#define KS4 4              // GEMM2: K=100 -> pad 128 -> 4 ksteps
#define NT4 7
#define KS2 4              // GEMM3: K=100 -> pad 128
#define NT2 24             // GEMM3: N=384 -> 24 tiles, exact
#define FRAG_HALVES 512    // 32 lanes * 16 halves per fragment
#define FB1 0
#define FB4 (FB1 + KS1*NT1)     // 14
#define FB2 (FB4 + KS4*NT4)     // 42
#define NFRAGS (FB2 + KS2*NT2)  // 138
#define TOT_HALVES (NFRAGS * FRAG_HALVES)  // 70656 halves = 141312 B in d_ws

// ---------------- main-kernel tiling ----------------
#define WAVES 4
#define ROWS_PER_BLOCK (WAVES * 16)
#define WBUF_BYTES 10240   // per-wave LDS: max(16x128 f16 staging=4KB, 16x160 f32 pass buf=10KB)

// ---------------- branch-free fast math (hardware v_exp_f32 / v_rcp_f32) --
__device__ inline float fast_exp(float x) { return __expf(x); }  // v_exp_f32 path

__device__ inline float fast_rcp(float x) { return __builtin_amdgcn_rcpf(x); }

// tanh(x) = 1 - 2/(exp(2x)+1): saturates to +/-1 for large |x| via exp inf/0,
// no branches, ~5 VALU ops.
__device__ inline float fast_tanh(float x) {
  float e = __expf(2.0f * x);
  return 1.0f - 2.0f * fast_rcp(e + 1.0f);
}

// Pack fp32 weights into f16, pre-swizzled into WMMA B-fragment order.
// B-fragment element mapping (derived from ISA 16-bit B layout):
//   lane l, half h (0..15):  K = kbase + (h&7) + 8*(l>>4) + 16*(h>>3),  N = nbase + (l&15)
__global__ void pack_weights(const float* __restrict__ W1,
                             const float* __restrict__ W4,
                             const float* __restrict__ W2,
                             _Float16* __restrict__ ws) {
  int e = blockIdx.x * blockDim.x + threadIdx.x;
  if (e >= TOT_HALVES) return;
  const float* W; int Kdim, Ndim, KS, rel;
  if (e < FB4 * FRAG_HALVES)      { W = W1; Kdim = K_D; Ndim = HID;  KS = KS1; rel = e; }
  else if (e < FB2 * FRAG_HALVES) { W = W4; Kdim = HID; Ndim = HID;  KS = KS4; rel = e - FB4 * FRAG_HALVES; }
  else                            { W = W2; Kdim = HID; Ndim = OUTN; KS = KS2; rel = e - FB2 * FRAG_HALVES; }
  int f    = rel >> 9;           // fragment within this weight, f = ntile*KS + kstep
  int lane = (rel >> 4) & 31;
  int h    = rel & 15;
  int ntile = f / KS;
  int kstep = f - ntile * KS;
  int k = kstep * 32 + (h & 7) + ((lane >> 4) << 3) + ((h >> 3) << 4);
  int n = ntile * 16 + (lane & 15);
  float v = (k < Kdim && n < Ndim) ? W[k * Ndim + n] : 0.0f;
  ws[e] = (_Float16)v;
}

// A-fragment (16-bit A, 16x32 layout): lane l holds row M=l&15,
//   halves 0..7  = K = kbase + 8*(l>>4) + 0..7
//   halves 8..15 = K = kbase + 16 + 8*(l>>4) + 0..7
__device__ inline v16h load_a(const _Float16* st, int lane, int kbase) {
  int row = lane & 15, hi = lane >> 4;
  const _Float16* p = st + row * 128 + kbase + hi * 8;
  v8h lo = *(const v8h*)(p);
  v8h hb = *(const v8h*)(p + 16);
  v16h r;
#pragma unroll
  for (int i = 0; i < 8; ++i) { r[i] = lo[i]; r[i + 8] = hb[i]; }
  return r;
}

__device__ inline v16h load_b(const _Float16* __restrict__ ws, int frag, int lane) {
  return *(const v16h*)(ws + (long long)frag * FRAG_HALVES + lane * 16);
}

__device__ inline v8f wmma_f16(v16h a, v16h b, v8f c) {
  return __builtin_amdgcn_wmma_f32_16x16x32_f16(false, a, false, b, (short)0, c, false, false);
}

__global__ __launch_bounds__(WAVES * 32)
void fused_mlp(const float* __restrict__ x,  const float* __restrict__ b1,
               const float* __restrict__ b4, const float* __restrict__ b2,
               const float* __restrict__ gum, const _Float16* __restrict__ ws,
               float* __restrict__ out) {
  __shared__ __align__(16) unsigned char smem[WAVES][WBUF_BYTES];
  const int wid  = threadIdx.x >> 5;
  const int lane = threadIdx.x & 31;
  const int l15  = lane & 15;
  const int hi   = lane >> 4;
  const long long rowBase = (long long)blockIdx.x * ROWS_PER_BLOCK + wid * 16;

  _Float16* st    = (_Float16*)&smem[wid][0];   // 16x128 f16, stride 128 halves
  float*    ob    = (float*)&smem[wid][0];      // f32 output staging (pass views)
  float*    rstat = (float*)&smem[wid][4096];   // [0..15]=rowmax, [16..31]=1/rowsum

  // ---- zero staging cols 112..127 (K padding for GEMM2/3 A fragments) ----
  {
    int row = lane >> 1, cb = 112 + (lane & 1) * 8;
    v8h z = {};
    *(v8h*)(st + row * 128 + cb) = z;
  }

  // ---- load x tile [16x64] f32 -> f16 staging cols 0..63 ----
  {
    int row = lane >> 1, cb = (lane & 1) * 32;
    const float* xp = x + (rowBase + row) * K_D + cb;
#pragma unroll
    for (int i = 0; i < 8; ++i) {
      v4f v = *(const v4f*)(xp + i * 4);
      v4h hq;
      hq[0] = (_Float16)v[0]; hq[1] = (_Float16)v[1];
      hq[2] = (_Float16)v[2]; hq[3] = (_Float16)v[3];
      *(v4h*)(st + row * 128 + cb + i * 4) = hq;
    }
  }

  // ================= GEMM1: [16x64] x [64x112] -> tanh -> staging =========
  v16h a10 = load_a(st, lane, 0);
  v16h a11 = load_a(st, lane, 32);
#pragma unroll
  for (int nt = 0; nt < NT1; ++nt) {
    v8f acc = {};
    acc = wmma_f16(a10, load_b(ws, FB1 + nt * KS1 + 0, lane), acc);
    acc = wmma_f16(a11, load_b(ws, FB1 + nt * KS1 + 1, lane), acc);
    int n = nt * 16 + l15;
    float bias = (n < HID) ? b1[n] : 0.0f;
#pragma unroll
    for (int r = 0; r < 8; ++r) {
      float hv = fast_tanh(acc[r] + bias);
      st[(r + hi * 8) * 128 + n] = (_Float16)hv;
    }
  }

  // ================= GEMM2: [16x128] x [128x112] -> tanh -> staging =======
  v16h a2[4];
#pragma unroll
  for (int ks = 0; ks < 4; ++ks) a2[ks] = load_a(st, lane, ks * 32);
#pragma unroll
  for (int nt = 0; nt < NT4; ++nt) {
    v8f acc = {};
#pragma unroll
    for (int ks = 0; ks < KS4; ++ks)
      acc = wmma_f16(a2[ks], load_b(ws, FB4 + nt * KS4 + ks, lane), acc);
    int n = nt * 16 + l15;
    float bias = (n < HID) ? b4[n] : 0.0f;
#pragma unroll
    for (int r = 0; r < 8; ++r) {
      float hv = fast_tanh(acc[r] + bias);
      st[(r + hi * 8) * 128 + n] = (_Float16)hv;
    }
  }

  // ---- preload GEMM3 A fragments before staging is reused as f32 buffer --
  v16h a3[4];
#pragma unroll
  for (int ks = 0; ks < 4; ++ks) a3[ks] = load_a(st, lane, ks * 32);

  // ================= GEMM3 pass A: tiles 0..3 -> row softmax (cols 0..63) =
#pragma unroll
  for (int nt = 0; nt < 4; ++nt) {
    v8f acc = {};
#pragma unroll
    for (int ks = 0; ks < KS2; ++ks)
      acc = wmma_f16(a3[ks], load_b(ws, FB2 + nt * KS2 + ks, lane), acc);
    int n = nt * 16 + l15;
    float bias = b2[n];
#pragma unroll
    for (int r = 0; r < 8; ++r)
      ob[(r + hi * 8) * 64 + n] = acc[r] + bias;
  }
  {
    int row = lane >> 1, cb = (lane & 1) * 32;
    float m = -__builtin_inff();
#pragma unroll
    for (int i = 0; i < 32; ++i) m = fmaxf(m, ob[row * 64 + cb + i]);
    m = fmaxf(m, __shfl_xor(m, 1, 32));
    float s = 0.0f;
#pragma unroll
    for (int i = 0; i < 32; ++i) s += fast_exp(ob[row * 64 + cb + i] - m);
    s += __shfl_xor(s, 1, 32);
    if ((lane & 1) == 0) { rstat[row] = m; rstat[16 + row] = fast_rcp(s); }
#pragma unroll
    for (int i = 0; i < 32; ++i) {           // coalesced softmax write-out
      int e = i * 32 + lane;
      int r2 = e >> 6, col = e & 63;
      float v = fast_exp(ob[r2 * 64 + col] - rstat[r2]) * rstat[16 + r2];
      out[(rowBase + r2) * OUTN + col] = v;
    }
  }

  // ===== GEMM3 passes B/C: tiles 4..13 / 14..23 -> gumbel straight-through
#pragma unroll 1
  for (int pass = 0; pass < 2; ++pass) {
    const int tile0 = 4 + pass * 10;
    const int gbase = pass * 32;               // group offset within 64 groups
#pragma unroll
    for (int nt = 0; nt < 10; ++nt) {
      int gnt = tile0 + nt;
      v8f acc = {};
#pragma unroll
      for (int ks = 0; ks < KS2; ++ks)
        acc = wmma_f16(a3[ks], load_b(ws, FB2 + gnt * KS2 + ks, lane), acc);
      int n = gnt * 16 + l15;
      float bias = b2[n];
#pragma unroll
      for (int r = 0; r < 8; ++r)
        ob[(r + hi * 8) * 160 + nt * 16 + l15] = acc[r] + bias;
    }
    // each lane owns one 5-wide modulation group across all 16 rows
#pragma unroll 1
    for (int row = 0; row < 16; ++row) {
      const float* gp = gum + ((rowBase + row) * (K_D * 5)) + (long long)(gbase + lane) * 5;
      float lv[5];
#pragma unroll
      for (int j = 0; j < 5; ++j)
        lv[j] = ob[row * 160 + lane * 5 + j] + gp[j];
      float m = lv[0]; int am = 0;
#pragma unroll
      for (int j = 1; j < 5; ++j)
        if (lv[j] > m) { m = lv[j]; am = j; }
      float ex[5], s = 0.0f;
#pragma unroll
      for (int j = 0; j < 5; ++j) { ex[j] = fast_exp(lv[j] - m); s += ex[j]; }
      float inv = fast_rcp(s);
      float* op = out + (rowBase + row) * OUTN + K_D + (long long)(gbase + lane) * 5;
#pragma unroll
      for (int j = 0; j < 5; ++j) {
        float ys = ex[j] * inv;
        float hard = (j == am) ? 1.0f : 0.0f;
        op[j] = ys + (hard - ys);              // straight-through, as in reference
      }
    }
  }
}

extern "C" void kernel_launch(void* const* d_in, const int* in_sizes, int n_in,
                              void* d_out, int out_size, void* d_ws, size_t ws_size,
                              hipStream_t stream) {
  const float* x   = (const float*)d_in[0];
  const float* W1  = (const float*)d_in[1];
  const float* b1  = (const float*)d_in[2];
  const float* W4  = (const float*)d_in[3];
  const float* b4  = (const float*)d_in[4];
  const float* W2  = (const float*)d_in[5];
  const float* b2  = (const float*)d_in[6];
  const float* gum = (const float*)d_in[7];
  _Float16* ws = (_Float16*)d_ws;
  float* out = (float*)d_out;

  // 1) repack weights fp32 -> f16 WMMA fragments in d_ws (deterministic, every call)
  {
    int t = 256, b = (TOT_HALVES + t - 1) / t;
    pack_weights<<<b, t, 0, stream>>>(W1, W4, W2, ws);
  }
  // 2) fused MLP + softmax + gumbel straight-through
  {
    int rows = in_sizes[0] / K_D;              // 262144
    int blocks = rows / ROWS_PER_BLOCK;        // 4096
    fused_mlp<<<blocks, WAVES * 32, 0, stream>>>(x, b1, b4, b2, gum, ws, out);
  }
}
